// Model_28209345200222
// MI455X (gfx1250) — compile-verified, matrix operand order checked by reading, not run on hardware
//
#include <hip/hip_runtime.h>
#include <hip/hip_bf16.h>

// ---------------------------------------------------------------------------
// Stacked FC -> 5x LSTM(48) -> 2x RNN(24) -> FC(2) over S=16384 timesteps.
// Phase 1 (time-parallel): fused fc1+fc2+lstm0 input projection via WMMA.
// Phase 2 (sequential):   8-wave systolic layer pipeline, weights in LDS.
// ---------------------------------------------------------------------------

#define S_LEN 16384
#define D_IN  6
#define D_FC  20
#define HL    48
#define G4    192   // 4*HL
#define HR    24
#define D_OUT 2

typedef __attribute__((ext_vector_type(16))) _Float16 v16h;
typedef __attribute__((ext_vector_type(8)))  float    v8f;

__device__ __forceinline__ float sigf(float x) {
    return 1.0f / (1.0f + __expf(-x));
}

// ===========================================================================
// Kernel 1: xW0[t][0:192] = relu(relu(x W1^T + b1) W2^T + b2) Wih0^T + (bih+bhh)
// 128 threads/block (4 waves), each wave owns one 16-row M tile.
// WMMA: A = 16x32 f16 (K=20 zero-padded), B = 32x16 f16, C/D = 16x16 f32.
// ===========================================================================
__global__ __launch_bounds__(128) void k_input_proj(
    const float* __restrict__ x,
    const float* __restrict__ fc1_W, const float* __restrict__ fc1_b,
    const float* __restrict__ fc2_W, const float* __restrict__ fc2_b,
    const float* __restrict__ Wih0,
    const float* __restrict__ bih0, const float* __restrict__ bhh0,
    float* __restrict__ xW0)
{
    __shared__ _Float16 Bsh[32 * G4];      // padded B: [k][n], 12 KB
    __shared__ _Float16 Ash[4][16][32];    // per-wave padded A tiles, 4 KB

    const int tid  = threadIdx.x;
    const int wave = tid >> 5;
    const int lane = tid & 31;
    const int rowBase = blockIdx.x * 64 + wave * 16;

    // Build padded B tile (shared across the whole block): Bsh[k][n] = Wih0[n][k]
    for (int idx = tid; idx < 32 * G4; idx += 128) {
        int k = idx / G4, n = idx % G4;
        Bsh[idx] = (k < D_FC) ? (_Float16)Wih0[n * D_FC + k] : (_Float16)0.0f;
    }

    // fc1 -> relu -> fc2 -> relu for this wave's 16 rows (lanes 0-15, one row each)
    if (lane < 16) {
        const int r = rowBase + lane;
        float xin[D_IN];
        #pragma unroll
        for (int i = 0; i < D_IN; ++i) xin[i] = x[r * D_IN + i];
        float h1[D_FC], h2[D_FC];
        #pragma unroll
        for (int j = 0; j < D_FC; ++j) {
            float a = fc1_b[j];
            #pragma unroll
            for (int i = 0; i < D_IN; ++i) a += xin[i] * fc1_W[j * D_IN + i];
            h1[j] = fmaxf(a, 0.0f);
        }
        #pragma unroll
        for (int j = 0; j < D_FC; ++j) {
            float a = fc2_b[j];
            #pragma unroll
            for (int i = 0; i < D_FC; ++i) a += h1[i] * fc2_W[j * D_FC + i];
            h2[j] = fmaxf(a, 0.0f);
        }
        #pragma unroll
        for (int k = 0; k < 32; ++k)
            Ash[wave][lane][k] = (k < D_FC) ? (_Float16)h2[k] : (_Float16)0.0f;
    }
    __syncthreads();

    // Pack the A fragment (16x32 f16 layout: lanes 0-15 K in {0..7,16..23},
    // lanes 16-31 K in {8..15,24..31}; VGPR v holds an adjacent K pair).
    const int m     = lane & 15;
    const int khalf = lane >> 4;
    v16h a;
    #pragma unroll
    for (int v = 0; v < 8; ++v) {
        int kb = 2 * (v & 3) + 16 * (v >> 2) + 8 * khalf;
        a[2 * v]     = Ash[wave][m][kb];
        a[2 * v + 1] = Ash[wave][m][kb + 1];
    }

    // 12 N-tiles of 16 gate columns each -> 12 v_wmma per wave.
    for (int nt = 0; nt < G4 / 16; ++nt) {
        const int ncol = nt * 16 + (lane & 15);
        v16h b;
        #pragma unroll
        for (int v = 0; v < 8; ++v) {
            int kb = 2 * (v & 3) + 16 * (v >> 2) + 8 * khalf;
            b[2 * v]     = Bsh[kb * G4 + ncol];
            b[2 * v + 1] = Bsh[(kb + 1) * G4 + ncol];
        }
        const float bias = bih0[ncol] + bhh0[ncol];   // bias folded into C
        v8f c;
        #pragma unroll
        for (int i = 0; i < 8; ++i) c[i] = bias;
        c = __builtin_amdgcn_wmma_f32_16x16x32_f16(
                false, a, false, b, (short)0, c, false, false);
        // C/D layout: lane = n + 16*(m>=8); VGPR i -> M = i + 8*(lane>>4)
        #pragma unroll
        for (int i = 0; i < 8; ++i) {
            int mm = i + 8 * khalf;
            xW0[(size_t)(rowBase + mm) * G4 + ncol] = c[i];
        }
    }
}

// ===========================================================================
// Kernel 2: sequential systolic pipeline. 1 block, 256 threads = 8 waves.
//   wave 0   : LSTM0 recurrence (xW0 precomputed)     wave 5 : RNN0 (48+24->24)
//   waves 1-4: LSTM1..4 (combined [Wih|Whh] 192x96)   wave 6 : RNN1 (24+24->24)
//                                                     wave 7 : relu + fc4 -> out
// All recurrent weights resident in LDS as f16 (~168 KB of the WGP's 320 KB).
// Hidden states double-buffered in LDS; one workgroup barrier per tick.
// ===========================================================================

// LDS element counts
#define N_W0   (G4 * HL)            // 9216  f16
#define N_WC   (4 * G4 * 96)        // 73728 f16
#define N_WR0  (HR * 72)            // 1728  f16
#define N_WR1  (HR * 48)            // 1152  f16
#define N_BSUM (4 * G4)             // 768   f32
#define LDS_BYTES ((N_W0 + N_WC + N_WR0 + N_WR1) * 2 + \
                   (N_BSUM + HR + HR + 48 + 2 + 8 * G4 + 8 * 2 * HL) * 4)

__global__ __launch_bounds__(256) void k_recurrent(
    const float* __restrict__ xW0,
    const float* __restrict__ W0hh_g,
    const float* __restrict__ WihL_g, const float* __restrict__ WhhL_g,
    const float* __restrict__ bihL_g, const float* __restrict__ bhhL_g,
    const float* __restrict__ r0Wih, const float* __restrict__ r0Whh,
    const float* __restrict__ r0bih, const float* __restrict__ r0bhh,
    const float* __restrict__ r1Wih, const float* __restrict__ r1Whh,
    const float* __restrict__ r1bih, const float* __restrict__ r1bhh,
    const float* __restrict__ fc4W, const float* __restrict__ fc4b,
    float* __restrict__ out)
{
    extern __shared__ char smem[];
    _Float16* W0  = (_Float16*)smem;       // LSTM0 Whh          [192][48]
    _Float16* WCb = W0  + N_W0;            // LSTM1-4 [Wih|Whh]  [4][192][96]
    _Float16* WR0 = WCb + N_WC;            // RNN0 [Wih|Whh]     [24][72]
    _Float16* WR1 = WR0 + N_WR0;           // RNN1 [Wih|Whh]     [24][48]
    float* BSUM = (float*)(WR1 + N_WR1);   // LSTM1-4 bih+bhh    [4][192]
    float* BR0  = BSUM + N_BSUM;           // [24]
    float* BR1  = BR0 + HR;                // [24]
    float* WF4  = BR1 + HR;                // fc4 weights        [2][24]
    float* BF4  = WF4 + 48;                // [2]
    float* GS   = BF4 + 2;                 // gate scratch       [8][192]
    float* HB   = GS + 8 * G4;             // hidden dbl-buffer  [8][2][48]

    const int tid = threadIdx.x;

    // ---- stage weights/biases into LDS (f16), zero hidden buffers ----
    for (int i = tid; i < N_W0; i += 256) W0[i] = (_Float16)W0hh_g[i];
    for (int i = tid; i < N_WC; i += 256) {
        int l = i / (G4 * 96), rem = i % (G4 * 96);
        int r = rem / 96, k = rem % 96;
        float w = (k < HL) ? WihL_g[((size_t)l * G4 + r) * HL + k]
                           : WhhL_g[((size_t)l * G4 + r) * HL + (k - HL)];
        WCb[i] = (_Float16)w;
    }
    for (int i = tid; i < N_WR0; i += 256) {
        int r = i / 72, k = i % 72;
        float w = (k < HL) ? r0Wih[r * HL + k] : r0Whh[r * HR + (k - HL)];
        WR0[i] = (_Float16)w;
    }
    for (int i = tid; i < N_WR1; i += 256) {
        int r = i / 48, k = i % 48;
        float w = (k < HR) ? r1Wih[r * HR + k] : r1Whh[r * HR + (k - HR)];
        WR1[i] = (_Float16)w;
    }
    for (int i = tid; i < N_BSUM; i += 256) BSUM[i] = bihL_g[i] + bhhL_g[i];
    for (int i = tid; i < HR; i += 256) { BR0[i] = r0bih[i] + r0bhh[i];
                                          BR1[i] = r1bih[i] + r1bhh[i]; }
    for (int i = tid; i < 48; i += 256) WF4[i] = fc4W[i];
    for (int i = tid; i < 2;  i += 256) BF4[i] = fc4b[i];
    for (int i = tid; i < 8 * 2 * HL; i += 256) HB[i] = 0.0f;
    __syncthreads();

    const int w    = tid >> 5;     // wave id == pipeline stage
    const int lane = tid & 31;
    float c0 = 0.0f, c1 = 0.0f;    // LSTM cell state (units lane, lane+32)
    float* gs = GS + w * G4;

    for (int tick = 0; tick < S_LEN + 7; ++tick) {
        const int t  = tick - w;           // this wave's timestep
        const int pw = tick & 1;           // write parity
        const int pr = pw ^ 1;             // read parity (produced last tick)

        if (t >= 0 && t < S_LEN) {
            if (w <= 4) {
                // ---------------- LSTM stage ----------------
                float acc[6];
                if (w == 0) {
                    if (lane == 0 && t + 32 < S_LEN)
                        __builtin_prefetch(&xW0[(size_t)(t + 32) * G4], 0, 0);
                    const float* hs = HB + pr * HL;              // self h
                    const float* xr = xW0 + (size_t)t * G4;
                    #pragma unroll
                    for (int m = 0; m < 6; ++m) acc[m] = xr[lane + 32 * m];
                    for (int k = 0; k < HL; ++k) {
                        float hv = hs[k];
                        #pragma unroll
                        for (int m = 0; m < 6; ++m)
                            acc[m] += hv * (float)W0[(lane + 32 * m) * HL + k];
                    }
                } else {
                    const float* hin = HB + ((w - 1) * 2 + pr) * HL;
                    const float* hs  = HB + (w * 2 + pr) * HL;
                    const _Float16* WL = WCb + (size_t)(w - 1) * G4 * 96;
                    const float* bs = BSUM + (w - 1) * G4;
                    #pragma unroll
                    for (int m = 0; m < 6; ++m) acc[m] = bs[lane + 32 * m];
                    for (int k = 0; k < HL; ++k) {
                        float hv = hin[k];
                        #pragma unroll
                        for (int m = 0; m < 6; ++m)
                            acc[m] += hv * (float)WL[(lane + 32 * m) * 96 + k];
                    }
                    for (int k = 0; k < HL; ++k) {
                        float hv = hs[k];
                        #pragma unroll
                        for (int m = 0; m < 6; ++m)
                            acc[m] += hv * (float)WL[(lane + 32 * m) * 96 + 48 + k];
                    }
                }
                #pragma unroll
                for (int m = 0; m < 6; ++m) gs[lane + 32 * m] = acc[m];
                // cross-lane gate exchange through LDS scratch
                asm volatile("s_wait_dscnt 0" ::: "memory");
                float* hw = HB + (w * 2 + pw) * HL;
                {
                    int u = lane;                       // units 0..31
                    float i_ = sigf(gs[u]);
                    float f_ = sigf(gs[HL + u]);
                    float g_ = tanhf(gs[2 * HL + u]);
                    float o_ = sigf(gs[3 * HL + u]);
                    c0 = f_ * c0 + i_ * g_;
                    hw[u] = o_ * tanhf(c0);
                }
                if (lane < 16) {                        // units 32..47
                    int u = 32 + lane;
                    float i_ = sigf(gs[u]);
                    float f_ = sigf(gs[HL + u]);
                    float g_ = tanhf(gs[2 * HL + u]);
                    float o_ = sigf(gs[3 * HL + u]);
                    c1 = f_ * c1 + i_ * g_;
                    hw[u] = o_ * tanhf(c1);
                }
            } else if (w == 5) {
                // ---------------- RNN0: 48(+24 self) -> 24 ----------------
                const float* hin = HB + (4 * 2 + pr) * HL;
                const float* hs  = HB + (5 * 2 + pr) * HL;
                if (lane < HR) {
                    float a = BR0[lane];
                    const _Float16* wr = WR0 + lane * 72;
                    for (int k = 0; k < HL; ++k) a += hin[k] * (float)wr[k];
                    for (int k = 0; k < HR; ++k) a += hs[k] * (float)wr[HL + k];
                    HB[(5 * 2 + pw) * HL + lane] = tanhf(a);
                }
            } else if (w == 6) {
                // ---------------- RNN1: 24(+24 self) -> 24 ----------------
                const float* hin = HB + (5 * 2 + pr) * HL;
                const float* hs  = HB + (6 * 2 + pr) * HL;
                if (lane < HR) {
                    float a = BR1[lane];
                    const _Float16* wr = WR1 + lane * 48;
                    for (int k = 0; k < HR; ++k) a += hin[k] * (float)wr[k];
                    for (int k = 0; k < HR; ++k) a += hs[k] * (float)wr[HR + k];
                    HB[(6 * 2 + pw) * HL + lane] = tanhf(a);
                }
            } else {
                // ---------------- relu + fc4 -> out ----------------
                const float* hin = HB + (6 * 2 + pr) * HL;
                if (lane < D_OUT) {
                    float a = BF4[lane];
                    for (int k = 0; k < HR; ++k)
                        a += fmaxf(hin[k], 0.0f) * WF4[lane * HR + k];
                    out[(size_t)t * D_OUT + lane] = a;
                }
            }
        }
        __syncthreads();   // one pipeline tick
    }
}

// ===========================================================================
extern "C" void kernel_launch(void* const* d_in, const int* in_sizes, int n_in,
                              void* d_out, int out_size, void* d_ws, size_t ws_size,
                              hipStream_t stream)
{
    const float* x        = (const float*)d_in[0];
    const float* fc1_W    = (const float*)d_in[1];
    const float* fc1_b    = (const float*)d_in[2];
    const float* fc2_W    = (const float*)d_in[3];
    const float* fc2_b    = (const float*)d_in[4];
    const float* l0_Wih   = (const float*)d_in[5];
    const float* l0_Whh   = (const float*)d_in[6];
    const float* l0_bih   = (const float*)d_in[7];
    const float* l0_bhh   = (const float*)d_in[8];
    const float* lr_Wih   = (const float*)d_in[9];
    const float* lr_Whh   = (const float*)d_in[10];
    const float* lr_bih   = (const float*)d_in[11];
    const float* lr_bhh   = (const float*)d_in[12];
    const float* r0_Wih   = (const float*)d_in[13];
    const float* r0_Whh   = (const float*)d_in[14];
    const float* r0_bih   = (const float*)d_in[15];
    const float* r0_bhh   = (const float*)d_in[16];
    const float* r1_Wih   = (const float*)d_in[17];
    const float* r1_Whh   = (const float*)d_in[18];
    const float* r1_bih   = (const float*)d_in[19];
    const float* r1_bhh   = (const float*)d_in[20];
    const float* fc4_W    = (const float*)d_in[21];
    const float* fc4_b    = (const float*)d_in[22];
    float* outp = (float*)d_out;
    float* xW0  = (float*)d_ws;   // [S_LEN][192] f32 (12.6 MB)

    // Phase 1: time-parallel WMMA input projection (256 blocks x 4 waves).
    k_input_proj<<<dim3(S_LEN / 64), dim3(128), 0, stream>>>(
        x, fc1_W, fc1_b, fc2_W, fc2_b, l0_Wih, l0_bih, l0_bhh, xW0);

    // Phase 2: single-WGP systolic recurrence with LDS-resident weights.
    k_recurrent<<<dim3(1), dim3(256), LDS_BYTES, stream>>>(
        xW0, l0_Whh, lr_Wih, lr_Whh, lr_bih, lr_bhh,
        r0_Wih, r0_Whh, r0_bih, r0_bhh,
        r1_Wih, r1_Whh, r1_bih, r1_bhh,
        fc4_W, fc4_b, outp);
}
